// FCOS_21973052686479
// MI455X (gfx1250) — matrix operand order repaired and checked by reading, compiled, not run on hardware
//
#include <hip/hip_runtime.h>
#include <stdint.h>
#include <stddef.h>

// ---------------------------------------------------------------------------
// Types for CDNA5 WMMA (wave32). bf16 A/B, f32 accumulate.
// ---------------------------------------------------------------------------
typedef __bf16 bf16;
typedef __attribute__((ext_vector_type(16))) __bf16 v16bf;
typedef __attribute__((ext_vector_type(8)))  __bf16 v8bf;
typedef __attribute__((ext_vector_type(4)))  __bf16 v4bf;
typedef __attribute__((ext_vector_type(8)))  float  v8f;

#define C_IN 256
// Concatenated output offsets (floats) for the 7-tuple, P = 2*5456 positions.
#define OFF_CLS  0
#define OFF_CS   21824
#define OFF_DXDY 76384
#define OFF_LR   109120
#define OFF_FID  130944
#define OFF_REG  141856
#define OFF_CTR  185504

__device__ __forceinline__ bf16 f2bf(float f) {
  union { float f; uint32_t u; } v; v.f = f;
  uint32_t r = v.u + 0x7FFFu + ((v.u >> 16) & 1u);   // round-to-nearest-even
  uint16_t h = (uint16_t)(r >> 16);
  return __builtin_bit_cast(bf16, h);
}
__device__ __forceinline__ float bf2f(bf16 b) {
  uint16_t h = __builtin_bit_cast(uint16_t, b);
  union { uint32_t u; float f; } v; v.u = ((uint32_t)h) << 16;
  return v.f;
}

// ---------------------------------------------------------------------------
// Layout packers
// ---------------------------------------------------------------------------
// NCHW f32 -> NHWC bf16 (channels contiguous = GEMM-K contiguous)
__global__ void pack_nchw_bf16(const float* __restrict__ src, bf16* __restrict__ dst,
                               int NM, int HW) {
  int i = blockIdx.x * 256 + threadIdx.x;
  if (i >= NM * C_IN) return;
  int c = i & 255;
  int p = i >> 8;
  int n = p / HW;
  int pp = p - n * HW;
  dst[i] = f2bf(src[((size_t)n * C_IN + c) * HW + pp]);
}

// OIHW f32 -> [kk][co][ci] bf16 (B fragment: contiguous ci per lane/column)
__global__ void pack_w_bf16(const float* __restrict__ w, bf16* __restrict__ wp) {
  int i = blockIdx.x * 256 + threadIdx.x;        // over 9*256*256
  if (i >= 9 * 256 * 256) return;
  int kk = i >> 16;
  int r  = i & 65535;
  int co = r >> 8;
  int ci = r & 255;
  wp[i] = f2bf(w[(size_t)(co * 256 + ci) * 9 + kk]);
}

// ---------------------------------------------------------------------------
// 3x3 SAME conv as implicit GEMM via v_wmma_f32_16x16x32_bf16.
// WG = 256 threads (8 waves). WG tile: M=128 positions x N=128 channels.
// Wave (wm,wn): rows [wm*32, +32), cols [wn*64, +64) -> 8 accumulators,
// 8 back-to-back WMMAs per 32-K step. K: 9 taps x 8 chunks = 72 steps.
// LDS is double-buffered: step s reads buf[s&1] while staging s+1 into
// buf[s&1^1]; one barrier per step; global loads issued before the WMMAs.
// ---------------------------------------------------------------------------
__global__ __launch_bounds__(256)
void conv3x3_wmma(const bf16* __restrict__ X, const bf16* __restrict__ Wp,
                  const float* __restrict__ bias, float* __restrict__ Y,
                  int NM, int HW, int H, int W) {
  constexpr int LDA = 40;                       // bf16 row stride (32 data + 8 pad)
  __shared__ bf16 shA[2][128 * LDA];            // 128 rows x 32 K activations
  __shared__ bf16 shB[2][128 * LDA];            // 128 cols x 32 K weights [co][ci]

  const int tid   = threadIdx.x;
  const int lane  = tid & 31;
  const int wv    = tid >> 5;
  const int wm    = wv & 3;                     // M sub-tile (32 rows)
  const int wn    = wv >> 2;                    // N sub-tile (64 cols)
  const int blkM  = blockIdx.x;
  const int cBase = blockIdx.y * 128;

  // Staging: thread t fills row/col (t>>1), 16-element chunk ((t&1)*16)
  const int ar = tid >> 1;
  const int aq = (tid & 1) * 16;
  const int ap = blkM * 128 + ar;               // output position staged by this thread
  int an = 0, ay = 0, ax = 0, rowOk = 0;
  if (ap < NM) { an = ap / HW; int r = ap - an * HW; ay = r / W; ax = r - ay * W; rowOk = 1; }
  const bf16* wBase = Wp + (size_t)(cBase + ar) * C_IN + aq;   // + kk*65536 + cb

  // Fragment read offsets (A: 8-elem chunks at K{0..7,16..23}/{8..15,24..31};
  // B: same 8-elem chunk K order as A, column = lane&15)
  const int frA = (wm * 32 + (lane & 15)) * LDA;
  const int k8  = (lane >> 4) * 8;
  const int frB = (wn * 64 + (lane & 15)) * LDA;
  const int k16 = (lane >> 4) * 16;

  v8f acc[2][4] = {};

  auto calcA = [&](int s, bool& av) -> const bf16* {
    int kk = s >> 3, cb = (s & 7) << 5;
    int q3 = (kk * 43) >> 7;                    // kk/3 for kk in [0,8]
    int iy = ay + q3 - 1, ix = ax + (kk - q3 * 3) - 1;
    av = rowOk && ((unsigned)iy < (unsigned)H) && ((unsigned)ix < (unsigned)W);
    return X + (size_t)(an * HW + iy * W + ix) * C_IN + cb + aq;
  };
  auto calcB = [&](int s) -> const bf16* {
    int kk = s >> 3, cb = (s & 7) << 5;
    return wBase + (size_t)kk * 65536 + cb;
  };

  { // prologue: stage step 0 into buffer 0
    bool av; const bf16* a = calcA(0, av);
    v8bf a0 = {}, a1 = {};
    if (av) { a0 = ((const v8bf*)a)[0]; a1 = ((const v8bf*)a)[1]; }
    const bf16* b = calcB(0);
    const v8bf b0 = ((const v8bf*)b)[0], b1 = ((const v8bf*)b)[1];
    *(v8bf*)&shA[0][ar * LDA + aq]     = a0;
    *(v8bf*)&shA[0][ar * LDA + aq + 8] = a1;
    *(v8bf*)&shB[0][ar * LDA + aq]     = b0;
    *(v8bf*)&shB[0][ar * LDA + aq + 8] = b1;
  }
  __syncthreads();

  for (int s = 0; s < 72; ++s) {
    const int cur = s & 1, nxt = cur ^ 1;
    const bool last = (s == 71);

    // Issue next step's global loads early (overlap with XDL work below)
    v8bf na0 = {}, na1 = {}, nb0 = {}, nb1 = {};
    if (!last) {
      bool av; const bf16* a = calcA(s + 1, av);
      if (av) { na0 = ((const v8bf*)a)[0]; na1 = ((const v8bf*)a)[1]; }
      const bf16* b = calcB(s + 1);
      nb0 = ((const v8bf*)b)[0];
      nb1 = ((const v8bf*)b)[1];
      if (s < 63) __builtin_prefetch(b + 65536, 0, 2);   // next tap's weights
    }

    // A fragments for both 16-row sub-tiles
    v16bf af[2];
#pragma unroll
    for (int ms = 0; ms < 2; ++ms) {
      const v8bf x0 = *(const v8bf*)&shA[cur][frA + ms * 16 * LDA + k8];
      const v8bf x1 = *(const v8bf*)&shA[cur][frA + ms * 16 * LDA + 16 + k8];
#pragma unroll
      for (int i = 0; i < 8; ++i) { af[ms][i] = x0[i]; af[ms][8 + i] = x1[i]; }
    }
    // 4 B fragments x 2 A fragments -> 8 independent WMMAs
#pragma unroll
    for (int nt = 0; nt < 4; ++nt) {
      const v8bf y0 = *(const v8bf*)&shB[cur][frB + nt * 16 * LDA + k8];
      const v8bf y1 = *(const v8bf*)&shB[cur][frB + nt * 16 * LDA + 16 + k8];
      v16bf bfr;
#pragma unroll
      for (int i = 0; i < 8; ++i) { bfr[i] = y0[i]; bfr[8 + i] = y1[i]; }
      acc[0][nt] = __builtin_amdgcn_wmma_f32_16x16x32_bf16(
          false, af[0], false, bfr, (short)0, acc[0][nt], false, false);
      acc[1][nt] = __builtin_amdgcn_wmma_f32_16x16x32_bf16(
          false, af[1], false, bfr, (short)0, acc[1][nt], false, false);
    }

    // Stage step s+1 into the other buffer, then one barrier
    if (!last) {
      *(v8bf*)&shA[nxt][ar * LDA + aq]     = na0;
      *(v8bf*)&shA[nxt][ar * LDA + aq + 8] = na1;
      *(v8bf*)&shB[nxt][ar * LDA + aq]     = nb0;
      *(v8bf*)&shB[nxt][ar * LDA + aq + 8] = nb1;
    }
    __syncthreads();
  }

  // C/D layout: lanes 0-15 VGPR v -> (M=v, N=lane); lanes 16-31 -> (M=v+8, N=lane-16)
  const int mrow = (lane >> 4) * 8;
  const int ncol = lane & 15;
#pragma unroll
  for (int ms = 0; ms < 2; ++ms) {
#pragma unroll
    for (int nt = 0; nt < 4; ++nt) {
      const int co = cBase + wn * 64 + nt * 16 + ncol;
      const float bv = bias[co];
#pragma unroll
      for (int v = 0; v < 8; ++v) {
        const int p = blkM * 128 + wm * 32 + ms * 16 + mrow + v;
        if (p < NM) Y[(size_t)p * C_IN + co] = acc[ms][nt][v] + bv;
      }
    }
  }
}

// ---------------------------------------------------------------------------
// GroupNorm: stats per (n, group) over 8 channels x HW, then apply+ReLU+bf16.
// ---------------------------------------------------------------------------
__global__ void gn_stats(const float* __restrict__ Y, float* __restrict__ stats, int HW) {
  __shared__ float s1[256], s2[256];
  const int b = blockIdx.x;             // n*32 + g
  const int n = b >> 5, g = b & 31;
  const float* base = Y + (size_t)n * HW * C_IN + g * 8;
  float a1 = 0.f, a2 = 0.f;
  for (int i = threadIdx.x; i < HW * 8; i += 256) {
    int p = i >> 3, c = i & 7;
    float v = base[(size_t)p * C_IN + c];
    a1 += v; a2 += v * v;
  }
  s1[threadIdx.x] = a1; s2[threadIdx.x] = a2;
  __syncthreads();
  for (int s = 128; s > 0; s >>= 1) {
    if (threadIdx.x < s) { s1[threadIdx.x] += s1[threadIdx.x + s]; s2[threadIdx.x] += s2[threadIdx.x + s]; }
    __syncthreads();
  }
  if (threadIdx.x == 0) {
    float cnt = (float)(HW * 8);
    float mu  = s1[0] / cnt;
    float var = s2[0] / cnt - mu * mu;
    stats[b * 2]     = mu;
    stats[b * 2 + 1] = rsqrtf(var + 1e-5f);
  }
}

__global__ void gn_apply_relu(const float* __restrict__ Y, const float* __restrict__ stats,
                              const float* __restrict__ gamma, const float* __restrict__ beta,
                              bf16* __restrict__ Xo, int NM, int HW) {
  int i = blockIdx.x * 256 + threadIdx.x;
  size_t base = (size_t)i * 4;
  if (base >= (size_t)NM * C_IN) return;
  int c = (int)(base & 255);
  int p = (int)(base >> 8);
  int n = p / HW;
  int g = c >> 3;                        // 4-chunk never crosses a group of 8
  float mu = stats[(n * 32 + g) * 2];
  float rs = stats[(n * 32 + g) * 2 + 1];
  const float4 v = *(const float4*)(Y + base);
  v4bf o;
  o[0] = f2bf(fmaxf((v.x - mu) * rs * gamma[c + 0] + beta[c + 0], 0.f));
  o[1] = f2bf(fmaxf((v.y - mu) * rs * gamma[c + 1] + beta[c + 1], 0.f));
  o[2] = f2bf(fmaxf((v.z - mu) * rs * gamma[c + 2] + beta[c + 2], 0.f));
  o[3] = f2bf(fmaxf((v.w - mu) * rs * gamma[c + 3] + beta[c + 3], 0.f));
  *(v4bf*)(Xo + base) = o;
}

// ---------------------------------------------------------------------------
// Head convs (tiny N) in scalar f32, fused with flatten/transpose, dxdy
// normalization, fid, and scatter into the concatenated output tuple.
// Activations read as b128 (8 x bf16) vectors; weight loads are wave-uniform.
// ---------------------------------------------------------------------------
__global__ void cls_heads(const bf16* __restrict__ T,
                          const float* __restrict__ cls_w, const float* __restrict__ cls_b,
                          const float* __restrict__ lr_w,  const float* __restrict__ lr_b,
                          const float* __restrict__ cs_w,  const float* __restrict__ cs_b,
                          const float* __restrict__ dx_w,  const float* __restrict__ dx_b,
                          float* __restrict__ out, int NM, int HW, int H, int W,
                          int lvlOff, int levelIdx) {
  int p = blockIdx.x * 256 + threadIdx.x;
  if (p >= NM) return;
  int n = p / HW; int r = p - n * HW; int y = r / W; int x = r - y * W;
  float a_cls0 = 0.f, a_cls1 = 0.f, a_lr0 = 0.f, a_lr1 = 0.f;
  float a_cs[5] = {0.f, 0.f, 0.f, 0.f, 0.f};
  float a_dx[3] = {0.f, 0.f, 0.f};
  for (int kk = 0; kk < 9; ++kk) {
    int iy = y + kk / 3 - 1, ix = x + kk % 3 - 1;
    if ((unsigned)iy >= (unsigned)H || (unsigned)ix >= (unsigned)W) continue;
    const bf16* t = T + (size_t)(n * HW + iy * W + ix) * C_IN;
    for (int ci = 0; ci < C_IN; ci += 8) {
      const v8bf tv = *(const v8bf*)(t + ci);
#pragma unroll
      for (int u = 0; u < 8; ++u) {
        float xv = bf2f(tv[u]);
        size_t wi = (size_t)(ci + u) * 9 + kk;    // OIHW flat: (o*256+ci)*9 + kk
        a_cls0 += xv * cls_w[wi];
        a_cls1 += xv * cls_w[2304 + wi];
        a_lr0  += xv * lr_w[wi];
        a_lr1  += xv * lr_w[2304 + wi];
#pragma unroll
        for (int j = 0; j < 5; ++j) a_cs[j] += xv * cs_w[(size_t)j * 2304 + wi];
#pragma unroll
        for (int j = 0; j < 3; ++j) a_dx[j] += xv * dx_w[(size_t)j * 2304 + wi];
      }
    }
  }
  size_t pos = (size_t)n * 5456 + lvlOff + r;
  out[OFF_CLS + pos * 2 + 0] = a_cls0 + cls_b[0];
  out[OFF_CLS + pos * 2 + 1] = a_cls1 + cls_b[1];
  out[OFF_LR + pos * 2 + 0]  = a_lr0 + lr_b[0];
  out[OFF_LR + pos * 2 + 1]  = a_lr1 + lr_b[1];
#pragma unroll
  for (int j = 0; j < 5; ++j) out[OFF_CS + pos * 5 + j] = a_cs[j] + cs_b[j];
  float d0 = fmaxf(a_dx[0] + dx_b[0], 0.f);
  float d1 = fmaxf(a_dx[1] + dx_b[1], 0.f);
  float d2 = fmaxf(a_dx[2] + dx_b[2], 0.f);
  float nrm = sqrtf(d1 * d1 + d2 * d2);
  float s = 0.1f / fmaxf(nrm, 1e-12f);
  out[OFF_DXDY + pos * 3 + 0] = d0;
  out[OFF_DXDY + pos * 3 + 1] = d1 * s;
  out[OFF_DXDY + pos * 3 + 2] = d2 * s;
  ((int*)out)[OFF_FID + pos] = levelIdx;        // int32 bits into the fid section
}

__global__ void reg_heads(const bf16* __restrict__ T,
                          const float* __restrict__ reg_w, const float* __restrict__ reg_b,
                          const float* __restrict__ ctr_w, const float* __restrict__ ctr_b,
                          float* __restrict__ out, int NM, int HW, int H, int W, int lvlOff) {
  int p = blockIdx.x * 256 + threadIdx.x;
  if (p >= NM) return;
  int n = p / HW; int r = p - n * HW; int y = r / W; int x = r - y * W;
  float a_reg[4] = {0.f, 0.f, 0.f, 0.f};
  float a_ctr = 0.f;
  for (int kk = 0; kk < 9; ++kk) {
    int iy = y + kk / 3 - 1, ix = x + kk % 3 - 1;
    if ((unsigned)iy >= (unsigned)H || (unsigned)ix >= (unsigned)W) continue;
    const bf16* t = T + (size_t)(n * HW + iy * W + ix) * C_IN;
    for (int ci = 0; ci < C_IN; ci += 8) {
      const v8bf tv = *(const v8bf*)(t + ci);
#pragma unroll
      for (int u = 0; u < 8; ++u) {
        float xv = bf2f(tv[u]);
        size_t wi = (size_t)(ci + u) * 9 + kk;
#pragma unroll
        for (int j = 0; j < 4; ++j) a_reg[j] += xv * reg_w[(size_t)j * 2304 + wi];
        a_ctr += xv * ctr_w[wi];
      }
    }
  }
  size_t pos = (size_t)n * 5456 + lvlOff + r;
#pragma unroll
  for (int j = 0; j < 4; ++j) out[OFF_REG + pos * 4 + j] = fmaxf(a_reg[j] + reg_b[j], 0.f);
  out[OFF_CTR + pos] = a_ctr + ctr_b[0];
}

// ---------------------------------------------------------------------------
// Host orchestration
// ---------------------------------------------------------------------------
extern "C" void kernel_launch(void* const* d_in, const int* in_sizes, int n_in,
                              void* d_out, int out_size, void* d_ws, size_t ws_size,
                              hipStream_t stream) {
  (void)n_in; (void)out_size; (void)ws_size;

  struct Params {
    const float* feat[5];
    const float *cls_w, *cls_b, *lr_w, *lr_b, *cs_w, *cs_b, *dx_w, *dx_b;
    const float *reg_w, *reg_b, *ctr_w, *ctr_b;
    const float *cw[4], *cb[4], *cg[4], *cgb[4];
    const float *rw[4], *rb[4], *rg[4], *rgb[4];
  } P;
  auto gp = [&](int i) { return (const float*)d_in[i]; };

  // Resolve input ordering from in_sizes (insertion order vs sorted pytree).
  const bool featsFirst = (in_sizes[0] == 2 * 256 * 64 * 64);
  if (featsFirst && in_sizes[5] == 9 * 256 * 256) {
    // Insertion order: feats, cls{tw(w,b,g,gb)x4, cls_w/b, lr_w/b, cs_w/b, dxdy_w/b}, reg{...}
    for (int i = 0; i < 5; ++i) P.feat[i] = gp(i);
    for (int l = 0; l < 4; ++l) { P.cw[l] = gp(5 + 4 * l); P.cb[l] = gp(6 + 4 * l); P.cg[l] = gp(7 + 4 * l); P.cgb[l] = gp(8 + 4 * l); }
    P.cls_w = gp(21); P.cls_b = gp(22); P.lr_w = gp(23); P.lr_b = gp(24);
    P.cs_w = gp(25);  P.cs_b = gp(26);  P.dx_w = gp(27); P.dx_b = gp(28);
    for (int l = 0; l < 4; ++l) { P.rw[l] = gp(29 + 4 * l); P.rb[l] = gp(30 + 4 * l); P.rg[l] = gp(31 + 4 * l); P.rgb[l] = gp(32 + 4 * l); }
    P.reg_w = gp(45); P.reg_b = gp(46); P.ctr_w = gp(47); P.ctr_b = gp(48);
  } else {
    // JAX sorted-key pytree order (with feats first or fully sorted).
    const int base = featsFirst ? 5 : 0;
    const int fb   = featsFirst ? 0 : 24;
    for (int i = 0; i < 5; ++i) P.feat[i] = gp(fb + i);
    P.cls_b = gp(base + 0); P.cls_w = gp(base + 1); P.cs_b = gp(base + 2); P.cs_w = gp(base + 3);
    P.dx_b  = gp(base + 4); P.dx_w  = gp(base + 5); P.lr_b = gp(base + 6); P.lr_w = gp(base + 7);
    for (int l = 0; l < 4; ++l) { P.cb[l] = gp(base + 8 + 4 * l); P.cg[l] = gp(base + 9 + 4 * l); P.cgb[l] = gp(base + 10 + 4 * l); P.cw[l] = gp(base + 11 + 4 * l); }
    P.ctr_b = gp(29); P.ctr_w = gp(30); P.reg_b = gp(31); P.reg_w = gp(32);
    for (int l = 0; l < 4; ++l) { P.rb[l] = gp(33 + 4 * l); P.rg[l] = gp(34 + 4 * l); P.rgb[l] = gp(35 + 4 * l); P.rw[l] = gp(36 + 4 * l); }
  }

  // Workspace carve (~29.1 MiB): Xin/Xa/Xb bf16 NHWC, Y f32, stats, packed weights.
  char* wsb = (char*)d_ws;
  bf16*  Xin   = (bf16*)(wsb);
  bf16*  Xa    = (bf16*)(wsb + ((size_t)4 << 20));
  bf16*  Xb    = (bf16*)(wsb + ((size_t)8 << 20));
  float* Ybuf  = (float*)(wsb + ((size_t)12 << 20));
  float* stats = (float*)(wsb + ((size_t)20 << 20));
  bf16* Wc[4]; bf16* Wr[4];
  size_t wp = ((size_t)20 << 20) + 65536;
  for (int l = 0; l < 4; ++l) { Wc[l] = (bf16*)(wsb + wp); wp += (size_t)9 * 256 * 256 * 2; }
  for (int l = 0; l < 4; ++l) { Wr[l] = (bf16*)(wsb + wp); wp += (size_t)9 * 256 * 256 * 2; }

  // Prepack tower weights to bf16 [kk][co][ci]
  for (int l = 0; l < 4; ++l) {
    pack_w_bf16<<<2304, 256, 0, stream>>>(P.cw[l], Wc[l]);
    pack_w_bf16<<<2304, 256, 0, stream>>>(P.rw[l], Wr[l]);
  }

  const int Wdim[5]   = {64, 32, 16, 8, 4};
  const int lvlOff[5] = {0, 4096, 5120, 5376, 5440};
  float* out = (float*)d_out;

  for (int lv = 0; lv < 5; ++lv) {
    const int H = Wdim[lv], Wd = Wdim[lv], HW = H * Wd, NM = 2 * HW;
    pack_nchw_bf16<<<(NM * C_IN + 255) / 256, 256, 0, stream>>>(P.feat[lv], Xin, NM, HW);

    auto tower = [&](bf16* const* Wpk, const float* const* tb, const float* const* tg,
                     const float* const* tgb) -> const bf16* {
      const bf16* src = Xin; bf16* dst = Xa;
      for (int l = 0; l < 4; ++l) {
        dim3 g((NM + 127) / 128, 2);   // N tile = 128 of 256 channels
        conv3x3_wmma<<<g, 256, 0, stream>>>(src, Wpk[l], tb[l], Ybuf, NM, HW, H, Wd);
        gn_stats<<<64, 256, 0, stream>>>(Ybuf, stats, HW);
        gn_apply_relu<<<(NM * 64 + 255) / 256, 256, 0, stream>>>(Ybuf, stats, tg[l], tgb[l], dst, NM, HW);
        src = dst; dst = (dst == Xa) ? Xb : Xa;
      }
      return src;
    };

    const bf16* Tc = tower(Wc, P.cb, P.cg, P.cgb);
    cls_heads<<<(NM + 255) / 256, 256, 0, stream>>>(Tc, P.cls_w, P.cls_b, P.lr_w, P.lr_b,
                                                    P.cs_w, P.cs_b, P.dx_w, P.dx_b,
                                                    out, NM, HW, H, Wd, lvlOff[lv], lv);
    const bf16* Tr = tower(Wr, P.rb, P.rg, P.rgb);
    reg_heads<<<(NM + 255) / 256, 256, 0, stream>>>(Tr, P.reg_w, P.reg_b, P.ctr_w, P.ctr_b,
                                                    out, NM, HW, H, Wd, lvlOff[lv]);
  }
}
